// GNNModule_69166153334815
// MI455X (gfx1250) — compile-verified, matrix operand order checked by reading, not run on hardware
//
#include <hip/hip_runtime.h>
#include <hip/hip_bf16.h>

#define NNODES   50000
#define NEDGES   300000
#define DH       256
#define DOUT     64
#define NLAYERS  3
#define NGRAPH   500
#define NGRAPH_P 512            // NGRAPH padded to 16
#define NEG_SLOPE 0.2f

typedef __attribute__((ext_vector_type(16))) __bf16 v16bf;
typedef __attribute__((ext_vector_type(8)))  __bf16 v8bf;
typedef __attribute__((ext_vector_type(2)))  __bf16 v2bf;
typedef __attribute__((ext_vector_type(8)))  float  v8f;

// ---------------------------------------------------------------- utilities
__global__ void fill_kernel(float* __restrict__ p, float v, int n) {
    int i = blockIdx.x * blockDim.x + threadIdx.x;
    if (i < n) p[i] = v;
}

__device__ __forceinline__ void atomicMaxF(float* addr, float val) {
    unsigned* ua = (unsigned*)addr;
    unsigned old = *ua;
    while (true) {
        float f = __uint_as_float(old);
        if (f >= val) break;
        unsigned assumed = old;
        old = atomicCAS(ua, assumed, __float_as_uint(val));
        if (old == assumed) break;
    }
}

// --------------------------------------------- f32 -> bf16 staging (padded)
// out[Mpad,K] bf16; rows >= M are zero so GEMM loads need no guards.
__global__ void to_bf16_pad(const float* __restrict__ in, __bf16* __restrict__ out,
                            int M, int Mpad, int K) {
    int i = blockIdx.x * blockDim.x + threadIdx.x;      // one thread per pair
    int pairsPerRow = K >> 1;
    if (i >= Mpad * pairsPerRow) return;
    int row = i / pairsPerRow;
    int col = (i - row * pairsPerRow) << 1;
    float2 v = (row < M) ? *(const float2*)(in + (size_t)row * K + col)
                         : make_float2(0.f, 0.f);
    v2bf pk;
    pk[0] = (__bf16)v.x;
    pk[1] = (__bf16)v.y;
    *(v2bf*)(out + (size_t)row * K + col) = pk;         // v_cvt_pk_bf16_f32 + b32 store
}

// ------------------------------------------------------- WMMA GEMM (X@W^T)
// Y[M,Nout] = X[M,K]_bf16 * (W[Nout,K]_bf16)^T (+bias), f32 accumulate.
// One wave computes a 16x64 strip: A fragment loaded once, 4 WMMAs per k-step.
// Fragment load = two global_load_b128 per matrix (16-bit layout: two
// contiguous 8-element runs at k0 + laneHi*8 and +16).
__global__ void wmma_gemm_bf16(const __bf16* __restrict__ X,
                               const __bf16* __restrict__ W,
                               const float* __restrict__ bias,
                               float* __restrict__ Y,
                               int M, int K, int Nout) {
    const int lane    = threadIdx.x & 31;
    const int wave    = (blockIdx.x * blockDim.x + threadIdx.x) >> 5;
    const int tilesN4 = Nout >> 6;                  // groups of 4 N-tiles
    const int tilesM  = (M + 15) >> 4;
    if (wave >= tilesM * tilesN4) return;
    const int tm  = wave / tilesN4;
    const int tn4 = wave - tm * tilesN4;
    const int laneLo = lane & 15;
    const int laneHi = lane >> 4;

    const __bf16* __restrict__ xrow = X + (size_t)(tm * 16 + laneLo) * K;
    const __bf16* __restrict__ wr0  = W + (size_t)(tn4 * 64 + laneLo) * K;

    v8f acc0 = {}, acc1 = {}, acc2 = {}, acc3 = {};
    for (int k0 = 0; k0 < K; k0 += 32) {
        const int base = k0 + laneHi * 8;
        v8bf alo = *(const v8bf*)(xrow + base);
        v8bf ahi = *(const v8bf*)(xrow + base + 16);
        v16bf a = __builtin_shufflevector(alo, ahi, 0,1,2,3,4,5,6,7,8,9,10,11,12,13,14,15);

        v8bf b0l = *(const v8bf*)(wr0 + base);
        v8bf b0h = *(const v8bf*)(wr0 + base + 16);
        v8bf b1l = *(const v8bf*)(wr0 + (size_t)16 * K + base);
        v8bf b1h = *(const v8bf*)(wr0 + (size_t)16 * K + base + 16);
        v8bf b2l = *(const v8bf*)(wr0 + (size_t)32 * K + base);
        v8bf b2h = *(const v8bf*)(wr0 + (size_t)32 * K + base + 16);
        v8bf b3l = *(const v8bf*)(wr0 + (size_t)48 * K + base);
        v8bf b3h = *(const v8bf*)(wr0 + (size_t)48 * K + base + 16);

        v16bf b0 = __builtin_shufflevector(b0l, b0h, 0,1,2,3,4,5,6,7,8,9,10,11,12,13,14,15);
        v16bf b1 = __builtin_shufflevector(b1l, b1h, 0,1,2,3,4,5,6,7,8,9,10,11,12,13,14,15);
        v16bf b2 = __builtin_shufflevector(b2l, b2h, 0,1,2,3,4,5,6,7,8,9,10,11,12,13,14,15);
        v16bf b3 = __builtin_shufflevector(b3l, b3h, 0,1,2,3,4,5,6,7,8,9,10,11,12,13,14,15);

        acc0 = __builtin_amdgcn_wmma_f32_16x16x32_bf16(false, a, false, b0, (short)0, acc0, false, false);
        acc1 = __builtin_amdgcn_wmma_f32_16x16x32_bf16(false, a, false, b1, (short)0, acc1, false, false);
        acc2 = __builtin_amdgcn_wmma_f32_16x16x32_bf16(false, a, false, b2, (short)0, acc2, false, false);
        acc3 = __builtin_amdgcn_wmma_f32_16x16x32_bf16(false, a, false, b3, (short)0, acc3, false, false);
    }

    // C/D layout: VGPR r -> row M = r + laneHi*8, col N = laneLo
#pragma unroll
    for (int j = 0; j < 4; ++j) {
        const v8f* accp = (j == 0) ? &acc0 : (j == 1) ? &acc1 : (j == 2) ? &acc2 : &acc3;
        const int ncol = tn4 * 64 + j * 16 + laneLo;
        const float bv = bias ? bias[ncol] : 0.f;
#pragma unroll
        for (int r = 0; r < 8; ++r) {
            int orow = tm * 16 + r + laneHi * 8;
            if (orow < M) Y[(size_t)orow * Nout + ncol] = (*accp)[r] + bv;
        }
    }
}

// --------------------------------------------- per-node attention dot products
// ai[n] = dot(H[n], att[0:D]);  aj[n] = dot(H[n], att[D:2D])
__global__ void node_att(const float* __restrict__ H, const float* __restrict__ att,
                         float* __restrict__ ai, float* __restrict__ aj, int N, int D) {
    int lane = threadIdx.x & 31;
    int node = (blockIdx.x * blockDim.x + threadIdx.x) >> 5;
    if (node >= N) return;
    const float* __restrict__ hrow = H + (size_t)node * D;
    float si = 0.f, sj = 0.f;
    for (int f = lane; f < D; f += 32) {
        float h = hrow[f];
        si += h * att[f];
        sj += h * att[D + f];
    }
#pragma unroll
    for (int off = 16; off > 0; off >>= 1) {
        si += __shfl_xor(si, off, 32);
        sj += __shfl_xor(sj, off, 32);
    }
    if (lane == 0) { ai[node] = si; aj[node] = sj; }
}

// ----------------------------------------------------------- edge pass 1
__global__ void edge_scores(const float* __restrict__ ai, const float* __restrict__ aj,
                            const int* __restrict__ src, const int* __restrict__ dst,
                            float* __restrict__ e, float* __restrict__ m, int E) {
    int i = blockIdx.x * blockDim.x + threadIdx.x;
    if (i >= E) return;
    float t = ai[dst[i]] + aj[src[i]];
    float v = t > 0.f ? t : NEG_SLOPE * t;      // leaky relu
    e[i] = v;
    atomicMaxF(&m[dst[i]], v);
}

// ----------------------------------------------------------- edge pass 2
__global__ void edge_exp(float* __restrict__ e, const float* __restrict__ m,
                         const int* __restrict__ dst, float* __restrict__ s, int E) {
    int i = blockIdx.x * blockDim.x + threadIdx.x;
    if (i >= E) return;
    int d = dst[i];
    float mm = m[d];
    mm = (mm > -1e37f) ? mm : 0.f;              // empty-segment guard
    float v = __expf(e[i] - mm);
    e[i] = v;                                    // overwrite with exp value
    atomicAdd(&s[d], v);
}

// ----------------------------------------------------------- edge pass 3
// One 256-thread block per edge: coalesced gather of H[src] row, coalesced
// atomic scatter into out[dst] row (L2-resident).
__global__ void edge_scatter(const float* __restrict__ ee, const float* __restrict__ s,
                             const float* __restrict__ H, const int* __restrict__ src,
                             const int* __restrict__ dst, float* __restrict__ out, int D) {
    int e = blockIdx.x;
    int f = threadIdx.x;
    int sn = src[e], dn = dst[e];
    float alpha = ee[e] / (s[dn] + 1e-16f);
    atomicAdd(&out[(size_t)dn * D + f], alpha * H[(size_t)sn * D + f]);
}

__global__ void add_bias(float* __restrict__ out, const float* __restrict__ b, int total, int D) {
    int i = blockIdx.x * blockDim.x + threadIdx.x;
    if (i < total) out[i] += b[i & (D - 1)];
}

// -------------------------------------------------- relu + global max pool
__global__ void relu_maxpool(const float* __restrict__ x, const int* __restrict__ batch,
                             float* __restrict__ g, int N, int D) {
    int i = blockIdx.x * blockDim.x + threadIdx.x;
    if (i >= N * D) return;
    int n = i / D, f = i & (D - 1);
    float v = x[i];
    v = v > 0.f ? v : 0.f;
    // nonnegative floats: integer order == float order, g pre-filled with 0.0f
    atomicMax((unsigned*)&g[(size_t)batch[n] * D + f], __float_as_uint(v));
}

// ------------------------------------------------------------- log-softmax
__global__ void log_softmax_kernel(float* __restrict__ io, int G, int C) {
    int lane = threadIdx.x & 31;
    int row  = (blockIdx.x * blockDim.x + threadIdx.x) >> 5;
    if (row >= G) return;
    float* __restrict__ r = io + (size_t)row * C;
    float mx = -1e30f;
    for (int c = lane; c < C; c += 32) mx = fmaxf(mx, r[c]);
#pragma unroll
    for (int off = 16; off > 0; off >>= 1) mx = fmaxf(mx, __shfl_xor(mx, off, 32));
    float sum = 0.f;
    for (int c = lane; c < C; c += 32) sum += __expf(r[c] - mx);
#pragma unroll
    for (int off = 16; off > 0; off >>= 1) sum += __shfl_xor(sum, off, 32);
    float lse = mx + __logf(sum);
    for (int c = lane; c < C; c += 32) r[c] = r[c] - lse;
}

// ---------------------------------------------------------------- launcher
extern "C" void kernel_launch(void* const* d_in, const int* in_sizes, int n_in,
                              void* d_out, int out_size, void* d_ws, size_t ws_size,
                              hipStream_t stream) {
    const float* x0    = (const float*)d_in[0];
    const int*   ei    = (const int*)  d_in[1];
    const int*   batch = (const int*)  d_in[2];
    const float* Ws    = (const float*)d_in[3];
    const float* atts  = (const float*)d_in[4];
    const float* bs    = (const float*)d_in[5];
    const float* pW1   = (const float*)d_in[6];
    const float* pb1   = (const float*)d_in[7];
    const float* pW2   = (const float*)d_in[8];
    const float* pb2   = (const float*)d_in[9];
    const int* src = ei;
    const int* dst = ei + NEDGES;

    char*  ws  = (char*)d_ws;
    size_t off = 0;
    auto alloc = [&](size_t bytes) -> void* {
        void* p = ws + off;
        off += (bytes + 255) & ~(size_t)255;
        return p;
    };
    float*  bufH = (float*)alloc(sizeof(float) * (size_t)NNODES * DH);
    float*  bufA = (float*)alloc(sizeof(float) * (size_t)NNODES * DH);
    float*  bufB = (float*)alloc(sizeof(float) * (size_t)NNODES * DH);
    __bf16* xb   = (__bf16*)alloc(sizeof(__bf16) * (size_t)NNODES * DH);    // bf16 X stage
    __bf16* wb   = (__bf16*)alloc(sizeof(__bf16) * (size_t)DH * DH);        // bf16 W stage
    __bf16* gb   = (__bf16*)alloc(sizeof(__bf16) * (size_t)NGRAPH_P * DH);  // bf16 pooled
    float*  ai   = (float*)alloc(sizeof(float) * NNODES);
    float*  aj   = (float*)alloc(sizeof(float) * NNODES);
    float*  mseg = (float*)alloc(sizeof(float) * NNODES);
    float*  sseg = (float*)alloc(sizeof(float) * NNODES);
    float*  ebuf = (float*)alloc(sizeof(float) * NEDGES);
    float*  g    = (float*)alloc(sizeof(float) * NGRAPH * DH);
    float*  hmid = (float*)alloc(sizeof(float) * NGRAPH * DH);

    auto cdiv = [](int a, int b) { return (a + b - 1) / b; };
    auto gemm = [&](const __bf16* Xb, const __bf16* Wb, const float* bias,
                    float* Y, int M, int Nout) {
        int waves = cdiv(M, 16) * (Nout >> 6);
        wmma_gemm_bf16<<<cdiv(waves * 32, 256), 256, 0, stream>>>(Xb, Wb, bias, Y, M, DH, Nout);
    };
    auto stage = [&](const float* in, __bf16* out, int M, int Mpad) {
        to_bf16_pad<<<cdiv(Mpad * (DH / 2), 256), 256, 0, stream>>>(in, out, M, Mpad, DH);
    };

    const float* X = x0;
    float* outs[NLAYERS] = { bufA, bufB, bufA };

    for (int l = 0; l < NLAYERS; ++l) {
        float* OUT = outs[l];
        fill_kernel<<<cdiv(NNODES, 256), 256, 0, stream>>>(mseg, -1e38f, NNODES);
        fill_kernel<<<cdiv(NNODES, 256), 256, 0, stream>>>(sseg, 0.f, NNODES);
        fill_kernel<<<cdiv(NNODES * DH, 256), 256, 0, stream>>>(OUT, 0.f, NNODES * DH);

        // stage operands in bf16, then H = X @ W^T on WMMA
        stage(X, xb, NNODES, NNODES);
        stage(Ws + (size_t)l * DH * DH, wb, DH, DH);
        gemm(xb, wb, nullptr, bufH, NNODES, DH);

        node_att<<<cdiv(NNODES * 32, 256), 256, 0, stream>>>(
            bufH, atts + (size_t)l * 2 * DH, ai, aj, NNODES, DH);

        edge_scores<<<cdiv(NEDGES, 256), 256, 0, stream>>>(ai, aj, src, dst, ebuf, mseg, NEDGES);
        edge_exp<<<cdiv(NEDGES, 256), 256, 0, stream>>>(ebuf, mseg, dst, sseg, NEDGES);
        edge_scatter<<<NEDGES, DH, 0, stream>>>(ebuf, sseg, bufH, src, dst, OUT, DH);

        add_bias<<<cdiv(NNODES * DH, 256), 256, 0, stream>>>(OUT, bs + (size_t)l * DH, NNODES * DH, DH);
        X = OUT;
    }

    // relu + global max pool
    fill_kernel<<<cdiv(NGRAPH * DH, 256), 256, 0, stream>>>(g, 0.f, NGRAPH * DH);
    relu_maxpool<<<cdiv(NNODES * DH, 256), 256, 0, stream>>>(X, batch, g, NNODES, DH);

    // post-MP MLP head (WMMA): hmid = g@pW1^T + pb1 ; logits = hmid@pW2^T + pb2
    stage(g, gb, NGRAPH, NGRAPH_P);
    stage(pW1, wb, DH, DH);
    gemm(gb, wb, pb1, hmid, NGRAPH, DH);

    stage(hmid, gb, NGRAPH, NGRAPH_P);
    stage(pW2, wb, DOUT, DOUT);
    gemm(gb, wb, pb2, (float*)d_out, NGRAPH, DOUT);

    log_softmax_kernel<<<cdiv(NGRAPH * 32, 256), 256, 0, stream>>>((float*)d_out, NGRAPH, DOUT);
}
